// EvoBinarizedLayerOptimized_35888746725578
// MI455X (gfx1250) — 8-bit matrix operand order repaired and checked by reading, compiled, not run on hardware
//
#include <hip/hip_runtime.h>
#include <hip/hip_bf16.h>
#include <stdint.h>

typedef int v8i __attribute__((ext_vector_type(8)));

#define POP        8
#define BATCH      512
#define IN_INTS    32
#define OUTF       2048

#define TILE_M     128
#define TILE_N     128
#define ASTRIDE    80
#define BSTRIDE    80

__device__ __forceinline__ unsigned int unpack4(unsigned int q) {
    return ((q & 0xFu) * 0x00204081u) & 0x01010101u;
}

__device__ __forceinline__ void stage_tiles(const long long* __restrict__ x,
                                            const long long* __restrict__ w,
                                            unsigned char* Ab, unsigned char* Bb,
                                            int m0, int n0, int p, int kw,
                                            int urow, int uhalf)
{
    unsigned long long xa = (unsigned long long)x[(size_t)(m0 + urow) * IN_INTS + kw];
    unsigned int h = (unsigned int)(xa >> (32 * uhalf));
    unsigned int a[8];
#pragma unroll
    for (int j = 0; j < 8; ++j) a[j] = unpack4(h >> (4 * j));
    int4* ad = (int4*)(Ab + urow * ASTRIDE + 32 * uhalf);
    ad[0] = make_int4((int)a[0], (int)a[1], (int)a[2], (int)a[3]);
    ad[1] = make_int4((int)a[4], (int)a[5], (int)a[6], (int)a[7]);

    const long long* wp = w + ((size_t)(p * IN_INTS + kw) * 2) * OUTF + (n0 + urow);
    unsigned long long w0 = (unsigned long long)wp[0];
    unsigned long long w1 = (unsigned long long)wp[OUTF];
    unsigned int h0 = (unsigned int)(w0 >> (32 * uhalf));
    unsigned int h1 = (unsigned int)(w1 >> (32 * uhalf));
    unsigned int d[8];
#pragma unroll
    for (int j = 0; j < 8; ++j) {
        unsigned int p0 = unpack4(h0 >> (4 * j));
        unsigned int p1 = unpack4(h1 >> (4 * j));
        d[j] = ((p0 | 0x80808080u) - p1) ^ 0x80808080u;
    }
    int4* bd = (int4*)(Bb + urow * BSTRIDE + 32 * uhalf);
    bd[0] = make_int4((int)d[0], (int)d[1], (int)d[2], (int)d[3]);
    bd[1] = make_int4((int)d[4], (int)d[5], (int)d[6], (int)d[7]);
}

__device__ __forceinline__ void compute_step(const unsigned char* Ab,
                                             const unsigned char* Bb,
                                             v8i (&acc)[4][2],
                                             int wm, int wn, int l16, int lhi)
{
    v8i bfrag[2];
#pragma unroll
    for (int ni = 0; ni < 2; ++ni) {
        const unsigned char* bp = Bb + (wn * 32 + ni * 16 + l16) * BSTRIDE + 16 * lhi;
        int4 b0 = *(const int4*)(bp);
        int4 b1 = *(const int4*)(bp + 32);
        bfrag[ni] = (v8i){b0.x, b0.y, b0.z, b0.w, b1.x, b1.y, b1.z, b1.w};
    }
#pragma unroll
    for (int mi = 0; mi < 4; ++mi) {
        const unsigned char* ap = Ab + (wm * 64 + mi * 16 + l16) * ASTRIDE + 16 * lhi;
        int2 a0 = *(const int2*)(ap);
        int2 a1 = *(const int2*)(ap + 8);
        int2 a2 = *(const int2*)(ap + 32);
        int2 a3 = *(const int2*)(ap + 40);
        v8i afrag = (v8i){a0.x, a0.y, a1.x, a1.y, a2.x, a2.y, a3.x, a3.y};
#pragma unroll
        for (int ni = 0; ni < 2; ++ni) {
            acc[mi][ni] = __builtin_amdgcn_wmma_i32_16x16x64_iu8(
                false, afrag, true, bfrag[ni], acc[mi][ni], false, false);
        }
    }
}

__global__ __launch_bounds__(256)
void evo_bin_linear_wmma(const long long* __restrict__ x,
                         const long long* __restrict__ w,
                         float* __restrict__ out)
{
    __shared__ __attribute__((aligned(16))) unsigned char Alds[2][TILE_M * ASTRIDE];
    __shared__ __attribute__((aligned(16))) unsigned char Blds[2][TILE_N * BSTRIDE];

    const int tid  = threadIdx.x;
    const int lane = tid & 31;
    const int wave = tid >> 5;

    const int n0 = blockIdx.x * TILE_N;
    const int m0 = blockIdx.y * TILE_M;
    const int p  = blockIdx.z;

    const int wm = wave & 1;
    const int wn = wave >> 1;
    const int l16 = lane & 15;
    const int lhi = lane >> 4;

    const int urow  = tid >> 1;
    const int uhalf = tid & 1;

    v8i acc[4][2] = {};

    stage_tiles(x, w, Alds[0], Blds[0], m0, n0, p, 0, urow, uhalf);
    __syncthreads();

    int buf = 0;
    for (int kw = 0; kw < IN_INTS - 1; ++kw) {
        stage_tiles(x, w, Alds[buf ^ 1], Blds[buf ^ 1], m0, n0, p, kw + 1, urow, uhalf);
        compute_step(Alds[buf], Blds[buf], acc, wm, wn, l16, lhi);
        __syncthreads();
        buf ^= 1;
    }
    compute_step(Alds[buf], Blds[buf], acc, wm, wn, l16, lhi);
    __syncthreads();

    int* biasLds = (int*)&Alds[0][0];
    if (tid < TILE_N) {
        int s = 0;
        for (int kw = 0; kw < IN_INTS; ++kw)
            s += __popcll((unsigned long long)
                     w[((size_t)(p * IN_INTS + kw) * 2 + 1) * OUTF + (n0 + tid)]);
        biasLds[tid] = s;
    }
    __syncthreads();

    int bias[2];
#pragma unroll
    for (int ni = 0; ni < 2; ++ni)
        bias[ni] = biasLds[wn * 32 + ni * 16 + l16];

#pragma unroll
    for (int mi = 0; mi < 4; ++mi) {
#pragma unroll
        for (int ni = 0; ni < 2; ++ni) {
            int o = n0 + wn * 32 + ni * 16 + l16;
#pragma unroll
            for (int r = 0; r < 8; ++r) {
                int m = m0 + wm * 64 + mi * 16 + r + 8 * lhi;
                out[((size_t)p * BATCH + m) * OUTF + o] = (float)(acc[mi][ni][r] + bias[ni]);
            }
        }
    }
}

extern "C" void kernel_launch(void* const* d_in, const int* in_sizes, int n_in,
                              void* d_out, int out_size, void* d_ws, size_t ws_size,
                              hipStream_t stream) {
    const long long* x = (const long long*)d_in[0];
    const long long* w = (const long long*)d_in[1];
    float* out = (float*)d_out;

    dim3 grid(OUTF / TILE_N, BATCH / TILE_M, POP);
    evo_bin_linear_wmma<<<grid, 256, 0, stream>>>(x, w, out);
}
